// PerformerModel_833223655432
// MI455X (gfx1250) — matrix operand order repaired and checked by reading, not compiled, not run on hardware
//
#include <hip/hip_runtime.h>
#include <hip/hip_bf16.h>

typedef __attribute__((ext_vector_type(16))) _Float16 v16h;
typedef __attribute__((ext_vector_type(8)))  _Float16 v8h;
typedef __attribute__((ext_vector_type(8)))  float    v8f;

constexpr int B = 64, S = 800, DIN = 64, D = 512, H = 8, L = 2, M = 64, C = 2;
constexpr int DH = 64, FF = 2048;
constexpr long BS = (long)B * S;                    // 51200
constexpr float NORM  = 0.35355339059327373f;      // DH^-0.25
constexpr float RATIO = 0.125f;                    // 1/sqrt(M)
constexpr float EPSK  = 1e-4f;
constexpr float LNEPS = 1e-5f;

enum { EPI_EMBED = 0, EPI_BIAS_H = 1, EPI_F32 = 2, EPI_RESID = 3,
       EPI_GELU_H = 4, EPI_DINV_H = 5 };

// ---------------------------------------------------------------------------
// Generic batched WMMA GEMM:  C[bat][Mrows x N] = A[bat] * B[bat] (+epilogue)
// A  : [Mrows x K] f16 row-major (leading dim lda)
// BT : B transposed, [N x K] f16 row-major (leading dim ldb) so both operand
//      fragments are contiguous 16B loads.
// Each wave computes a 32 x 64 strip: 2 row-subtiles x 4 col-tiles = 8 WMMA
// accumulators (64 VGPRs). B fragments are register-reused across both row
// subtiles; A fragments across the 4 column tiles -> 8 WMMAs per 12 b128
// loads per 32-wide k-step.  ~160 VGPRs incl. pipelined loads; launch_bounds
// (256,1) lifts the 128-VGPR occupancy cap so nothing spills.
// Batch index decomposes as bat = batO*Hdiv + batI with independent strides.
// ---------------------------------------------------------------------------
template <int MT>
__global__ __launch_bounds__(256, 1) void performer_gemm_wmma(
    const _Float16* __restrict__ A, const _Float16* __restrict__ BT,
    const float* __restrict__ bias, float* __restrict__ outF,
    _Float16* __restrict__ outH, const float* __restrict__ resid,
    const float* __restrict__ posp, const float* __restrict__ dinvp,
    int Mrows, int N, int K, int lda, int ldb, int ldc,
    int nbat, int Hdiv,
    long ao, long ai, long bo_, long bi_, long co, long ci,
    int Sdim, int epi)
{
    int wid  = (int)((blockIdx.x * blockDim.x + threadIdx.x) >> 5);
    int lane = threadIdx.x & 31;
    int tiles_n = N >> 6;
    int tiles_m = Mrows / (16 * MT);
    int per_bat = tiles_m * tiles_n;
    if (wid >= nbat * per_bat) return;          // wave-uniform: EXEC stays full

    int bat = wid / per_bat;
    int rem = wid - bat * per_bat;
    int tm  = rem / tiles_n;
    int tn  = rem - tm * tiles_n;
    int batO = bat / Hdiv;
    int batI = bat - batO * Hdiv;

    const _Float16* Ab = A  + (long)batO * ao  + (long)batI * ai
                            + (long)(tm * 16 * MT) * lda;
    const _Float16* Bb = BT + (long)batO * bo_ + (long)batI * bi_
                            + (long)(tn << 6) * ldb;

    int hs = lane >> 4;      // half-wave select
    int rc = lane & 15;      // row (A/C) or column (B/C)

    // A frag (16x32 f16): lanes 0-15 row=rc hold K=lo..lo+7 and lo+16..lo+23
    // with lo = 8*hs.  B frag (32x16 f16): lane col=rc holds K=lo..lo+7 and lo+16..lo+23.
    const _Float16* ap = Ab + rc * lda + hs * 8;
    const _Float16* bp = Bb + rc * ldb + hs * 8;

    v8f acc[MT][4];
#pragma unroll
    for (int mt = 0; mt < MT; ++mt)
#pragma unroll
        for (int t = 0; t < 4; ++t)
#pragma unroll
            for (int e = 0; e < 8; ++e) acc[mt][t][e] = 0.0f;

    for (int k0 = 0; k0 < K; k0 += 32) {
        v16h af[MT];
#pragma unroll
        for (int mt = 0; mt < MT; ++mt) {
            const _Float16* amt = ap + (long)mt * 16 * lda + k0;
            v8h a0 = *(const v8h*)(amt);
            v8h a1 = *(const v8h*)(amt + 16);
#pragma unroll
            for (int i = 0; i < 8; ++i) { af[mt][i] = a0[i]; af[mt][i + 8] = a1[i]; }
        }
#pragma unroll
        for (int t = 0; t < 4; ++t) {
            const _Float16* bpt = bp + (long)t * 16 * ldb + k0;
            v8h b0 = *(const v8h*)(bpt);
            v8h b1 = *(const v8h*)(bpt + 16);
            v16h bf;
#pragma unroll
            for (int i = 0; i < 8; ++i) { bf[i] = b0[i]; bf[i + 8] = b1[i]; }
#pragma unroll
            for (int mt = 0; mt < MT; ++mt)
                acc[mt][t] = __builtin_amdgcn_wmma_f32_16x16x32_f16(
                    false, af[mt], false, bf, (short)0, acc[mt][t], false, false);
        }
    }

    // C layout: reg e, lane -> row = e + 8*hs, col = rc
    long cbase = (long)batO * co + (long)batI * ci;
#pragma unroll
    for (int mt = 0; mt < MT; ++mt) {
#pragma unroll
        for (int t = 0; t < 4; ++t) {
            int gc = (tn << 6) + (t << 4) + rc;
#pragma unroll
            for (int e = 0; e < 8; ++e) {
                int gr = tm * 16 * MT + mt * 16 + e + (hs << 3);
                float vacc = acc[mt][t][e];
                long cidx = cbase + (long)gr * ldc + gc;
                if (epi == EPI_EMBED) {
                    int s = gr % Sdim;
                    outF[cidx] = vacc + bias[gc] + posp[(long)s * N + gc];
                } else if (epi == EPI_BIAS_H) {
                    outH[cidx] = (_Float16)(vacc + bias[gc]);
                } else if (epi == EPI_F32) {
                    outF[cidx] = vacc;
                } else if (epi == EPI_RESID) {
                    outF[cidx] = resid[cidx] + vacc + bias[gc];
                } else if (epi == EPI_GELU_H) {
                    float xg = vacc + bias[gc];
                    float inner = 0.7978845608028654f * (xg + 0.044715f * xg * xg * xg);
                    outH[cidx] = (_Float16)(0.5f * xg * (1.0f + tanhf(inner)));
                } else { // EPI_DINV_H
                    float sc = dinvp[(long)bat * Sdim + gr];
                    outH[cidx] = (_Float16)(vacc * sc);
                }
            }
        }
    }
}

// --------------------------- small helper kernels --------------------------

__global__ void k_conv_scale(const float* __restrict__ s, _Float16* __restrict__ d,
                             long n, float sc) {
    long i = (long)blockIdx.x * blockDim.x + threadIdx.x;
    if (i < n) d[i] = (_Float16)(s[i] * sc);
}

__global__ void k_transpose(const float* __restrict__ s, _Float16* __restrict__ d,
                            int R, int Cc) {
    long i = (long)blockIdx.x * blockDim.x + threadIdx.x;
    if (i < (long)R * Cc) {
        long r = i / Cc;
        long c = i - r * Cc;
        d[c * R + r] = (_Float16)s[i];
    }
}

// LayerNorm: one wave per row of D=512, shuffle reduction, f16 output.
__global__ void k_layernorm(const float* __restrict__ x, const float* __restrict__ g,
                            const float* __restrict__ b, _Float16* __restrict__ y,
                            int rows) {
    int row = blockIdx.x * (blockDim.x >> 5) + (threadIdx.x >> 5);
    if (row >= rows) return;
    int lane = threadIdx.x & 31;
    const float* xr = x + (long)row * D;
    float s1 = 0.f, s2 = 0.f;
    for (int i = lane; i < D; i += 32) { float v = xr[i]; s1 += v; s2 += v * v; }
    for (int o = 16; o > 0; o >>= 1) {
        s1 += __shfl_xor(s1, o, 32);
        s2 += __shfl_xor(s2, o, 32);
    }
    float mean = s1 * (1.0f / D);
    float var  = s2 * (1.0f / D) - mean * mean;
    float rstd = rsqrtf(var + LNEPS);
    _Float16* yr = y + (long)row * D;
    for (int i = lane; i < D; i += 32)
        yr[i] = (_Float16)((xr[i] - mean) * rstd * g[i] + b[i]);
}

// Query features: per (b,h,s) row-max stabilizer + diag + exp.
__global__ void k_qp(const float* __restrict__ dd, const _Float16* __restrict__ q,
                     _Float16* __restrict__ qp, int total) {
    int i = blockIdx.x * blockDim.x + threadIdx.x;
    if (i >= total) return;
    int bh = i / S, s = i - bh * S;
    int b = bh >> 3, h = bh & 7;                       // H == 8
    const float* ddr = dd + (long)i * M;
    const _Float16* qr = q + ((long)b * S + s) * D + h * DH;
    float mx = -1e30f;
    for (int m = 0; m < M; ++m) mx = fmaxf(mx, ddr[m]);
    float sq = 0.f;
    for (int d0 = 0; d0 < DH; ++d0) { float v = (float)qr[d0]; sq += v * v; }
    float diag = 0.0625f * sq;                         // 0.5 * NORM^2
    _Float16* o = qp + (long)i * M;
    for (int m = 0; m < M; ++m)
        o[m] = (_Float16)(RATIO * (expf(ddr[m] - diag - mx) + EPSK));
}

__global__ void k_rowmax(const float* __restrict__ dd, float* __restrict__ rmax,
                         int total) {
    int i = blockIdx.x * blockDim.x + threadIdx.x;
    if (i >= total) return;
    const float* ddr = dd + (long)i * M;
    float mx = -1e30f;
    for (int m = 0; m < M; ++m) mx = fmaxf(mx, ddr[m]);
    rmax[i] = mx;
}

__global__ void k_stab(const float* __restrict__ rmax, float* __restrict__ stab) {
    __shared__ float sm[256];
    int bh = blockIdx.x;
    float mx = -1e30f;
    for (int s = threadIdx.x; s < S; s += 256) mx = fmaxf(mx, rmax[(long)bh * S + s]);
    sm[threadIdx.x] = mx;
    __syncthreads();
    for (int o = 128; o > 0; o >>= 1) {
        if ((int)threadIdx.x < o) sm[threadIdx.x] = fmaxf(sm[threadIdx.x], sm[threadIdx.x + o]);
        __syncthreads();
    }
    if (threadIdx.x == 0) stab[bh] = sm[0];
}

// Key features: global (per b,h) stabilizer.
__global__ void k_kp(const float* __restrict__ dd, const _Float16* __restrict__ kk,
                     const float* __restrict__ stab, _Float16* __restrict__ kp,
                     int total) {
    int i = blockIdx.x * blockDim.x + threadIdx.x;
    if (i >= total) return;
    int bh = i / S, s = i - bh * S;
    int b = bh >> 3, h = bh & 7;
    const float* ddr = dd + (long)i * M;
    const _Float16* kr = kk + ((long)b * S + s) * D + h * DH;
    float sq = 0.f;
    for (int d0 = 0; d0 < DH; ++d0) { float v = (float)kr[d0]; sq += v * v; }
    float diag = 0.0625f * sq;
    float st = stab[bh];
    _Float16* o = kp + (long)i * M;
    for (int m = 0; m < M; ++m)
        o[m] = (_Float16)(RATIO * (expf(ddr[m] - diag - st) + EPSK));
}

__global__ void k_ksum(const _Float16* __restrict__ kp, float* __restrict__ ksum,
                       int total) {
    int i = blockIdx.x * blockDim.x + threadIdx.x;
    if (i >= total) return;
    int bh = i >> 6, m = i & 63;
    const _Float16* base = kp + (long)bh * S * M + m;
    float sv = 0.f;
    for (int s = 0; s < S; ++s) sv += (float)base[(long)s * M];
    ksum[i] = sv;
}

// ctxT[b,h,d,m] = sum_s kp[b,h,s,m] * v[b,s,h*DH+d]   (stored transposed = B^T)
__global__ void k_ctx(const _Float16* __restrict__ kp, const _Float16* __restrict__ v,
                      _Float16* __restrict__ ctxT, int total) {
    int i = blockIdx.x * blockDim.x + threadIdx.x;
    if (i >= total) return;
    int m = i & 63;
    int t = i >> 6;
    int d0 = t & 63;
    int bh = t >> 6;
    int b = bh >> 3, h = bh & 7;
    const _Float16* kpb = kp + (long)bh * S * M + m;
    const _Float16* vb  = v + ((long)b * S) * D + h * DH + d0;
    float acc = 0.f;
    for (int s = 0; s < S; ++s)
        acc += (float)kpb[(long)s * M] * (float)vb[(long)s * D];
    ctxT[((long)bh * DH + d0) * M + m] = (_Float16)acc;
}

__global__ void k_dinv(const _Float16* __restrict__ qp, const float* __restrict__ ksum,
                       float* __restrict__ dinv, int total) {
    int i = blockIdx.x * blockDim.x + threadIdx.x;
    if (i >= total) return;
    int bh = i / S;
    const _Float16* qr = qp + (long)i * M;
    const float* ks = ksum + (long)bh * M;
    float sv = 0.f;
    for (int m = 0; m < M; ++m) sv += (float)qr[m] * ks[m];
    dinv[i] = 1.0f / sv;
}

__global__ void k_hbar(const float* __restrict__ h, float* __restrict__ hbar,
                       int total) {
    int i = blockIdx.x * blockDim.x + threadIdx.x;
    if (i >= total) return;
    int b = i / D, d0 = i - b * D;
    const float* hb = h + ((long)b * S) * D + d0;
    float sv = 0.f;
    for (int s = 0; s < S; ++s) sv += hb[(long)s * D];
    hbar[i] = sv * (1.0f / S);
}

__global__ void k_fc(const float* __restrict__ hbar, const float* __restrict__ fcw,
                     const float* __restrict__ fcb, float* __restrict__ out, int total) {
    int i = blockIdx.x * blockDim.x + threadIdx.x;
    if (i >= total) return;
    int b = i / C, c = i - b * C;
    float sv = fcb[c];
    for (int d0 = 0; d0 < D; ++d0) sv += hbar[(long)b * D + d0] * fcw[(long)d0 * C + c];
    out[i] = sv;
}

// ------------------------------- host side ---------------------------------

static inline unsigned nblk(long n, int bs) { return (unsigned)((n + bs - 1) / bs); }

static void gemm(hipStream_t st, const _Float16* A, const _Float16* BT,
                 const float* bias, float* oF, _Float16* oH, const float* resid,
                 const float* posp, const float* dinvp,
                 int Mr, int N, int K, int lda, int ldb, int ldc,
                 int nbat, int Hdiv, long ao, long ai, long bo_, long bi_,
                 long co, long ci, int Sdim, int epi)
{
    constexpr int MT = 2;                       // 32-row strips: 51200%32==0, 800%32==0
    long waves = (long)nbat * (Mr / (16 * MT)) * (N >> 6);
    long thr = waves * 32;
    performer_gemm_wmma<MT><<<dim3(nblk(thr, 256)), dim3(256), 0, st>>>(
        A, BT, bias, oF, oH, resid, posp, dinvp,
        Mr, N, K, lda, ldb, ldc, nbat, Hdiv, ao, ai, bo_, bi_, co, ci, Sdim, epi);
}

extern "C" void kernel_launch(void* const* d_in, const int* in_sizes, int n_in,
                              void* d_out, int out_size, void* d_ws, size_t ws_size,
                              hipStream_t stream)
{
    const float* x     = (const float*)d_in[0];
    const float* emb_w = (const float*)d_in[1];
    const float* emb_b = (const float*)d_in[2];
    const float* pos   = (const float*)d_in[3];
    const float* ln1_g = (const float*)d_in[4];
    const float* ln1_b = (const float*)d_in[5];
    const float* wq    = (const float*)d_in[6];
    const float* bq    = (const float*)d_in[7];
    const float* wk    = (const float*)d_in[8];
    const float* bk    = (const float*)d_in[9];
    const float* wv    = (const float*)d_in[10];
    const float* bv    = (const float*)d_in[11];
    const float* wo    = (const float*)d_in[12];
    const float* bo    = (const float*)d_in[13];
    const float* proj  = (const float*)d_in[14];
    const float* ln2_g = (const float*)d_in[15];
    const float* ln2_b = (const float*)d_in[16];
    const float* w1    = (const float*)d_in[17];
    const float* b1    = (const float*)d_in[18];
    const float* w2    = (const float*)d_in[19];
    const float* b2    = (const float*)d_in[20];
    const float* fc_w  = (const float*)d_in[21];
    const float* fc_b  = (const float*)d_in[22];
    float* out = (float*)d_out;

    // ---- workspace layout (recomputed deterministically every call) ----
    char* ws = (char*)d_ws;
    size_t off = 0;
    auto take = [&](size_t bytes) -> char* {
        char* p = ws + off;
        off = (off + bytes + 255) & ~(size_t)255;
        return p;
    };

    _Float16* xh   = (_Float16*)take((size_t)BS * DIN * 2);
    _Float16* embT = (_Float16*)take((size_t)D * DIN * 2);
    _Float16 *wqT[L], *wkT[L], *wvT[L], *woT[L], *w1T[L], *w2T[L], *projh[L];
    for (int l = 0; l < L; ++l) {
        wqT[l]   = (_Float16*)take((size_t)D * D * 2);
        wkT[l]   = (_Float16*)take((size_t)D * D * 2);
        wvT[l]   = (_Float16*)take((size_t)D * D * 2);
        woT[l]   = (_Float16*)take((size_t)D * D * 2);
        w1T[l]   = (_Float16*)take((size_t)FF * D * 2);   // [FF x D]
        w2T[l]   = (_Float16*)take((size_t)D * FF * 2);   // [D x FF]
        projh[l] = (_Float16*)take((size_t)M * DH * 2);
    }
    float*    h   = (float*)   take((size_t)BS * D * 4);
    _Float16* y   = (_Float16*)take((size_t)BS * D * 2);
    _Float16* att = (_Float16*)take((size_t)BS * D * 2);
    // q,k,v,dd contiguous; ffn1 (BS*FF f16 = 210MB) aliases this 262MB region
    char* qregion = ws + off;
    _Float16* q  = (_Float16*)take((size_t)BS * D * 2);
    _Float16* kk = (_Float16*)take((size_t)BS * D * 2);
    _Float16* v  = (_Float16*)take((size_t)BS * D * 2);
    float*    dd = (float*)   take((size_t)B * H * S * M * 4);
    _Float16* ffn1 = (_Float16*)qregion;
    _Float16* qp = (_Float16*)take((size_t)B * H * S * M * 2);
    _Float16* kp = (_Float16*)take((size_t)B * H * S * M * 2);
    float* rmax = (float*)take((size_t)B * H * S * 4);
    float* stab = (float*)take((size_t)B * H * 4);
    float* ksum = (float*)take((size_t)B * H * M * 4);
    _Float16* ctxT = (_Float16*)take((size_t)B * H * DH * M * 2);
    float* dinv = (float*)take((size_t)B * H * S * 4);
    float* hbar = (float*)take((size_t)B * D * 4);
    (void)ws_size; (void)in_sizes; (void)n_in; (void)out_size;

    const int BHS = B * H * S;

    // ---- weight prep: f16 conversion + transpose (B^T operand form) ----
    k_conv_scale<<<nblk(BS * DIN, 256), 256, 0, stream>>>(x, xh, BS * DIN, 1.0f);
    k_transpose<<<nblk((long)DIN * D, 256), 256, 0, stream>>>(emb_w, embT, DIN, D);
    for (int l = 0; l < L; ++l) {
        k_transpose<<<nblk((long)D * D, 256), 256, 0, stream>>>(wq + (long)l * D * D, wqT[l], D, D);
        k_transpose<<<nblk((long)D * D, 256), 256, 0, stream>>>(wk + (long)l * D * D, wkT[l], D, D);
        k_transpose<<<nblk((long)D * D, 256), 256, 0, stream>>>(wv + (long)l * D * D, wvT[l], D, D);
        k_transpose<<<nblk((long)D * D, 256), 256, 0, stream>>>(wo + (long)l * D * D, woT[l], D, D);
        k_transpose<<<nblk((long)D * FF, 256), 256, 0, stream>>>(w1 + (long)l * D * FF, w1T[l], D, FF);
        k_transpose<<<nblk((long)FF * D, 256), 256, 0, stream>>>(w2 + (long)l * FF * D, w2T[l], FF, D);
        k_conv_scale<<<nblk((long)M * DH, 256), 256, 0, stream>>>(proj + (long)l * M * DH, projh[l],
                                                                  (long)M * DH, NORM);
    }

    // ---- embedding: h = x @ emb_w + emb_b + pos ----
    gemm(stream, xh, embT, emb_b, h, nullptr, nullptr, pos, nullptr,
         (int)BS, D, DIN, DIN, DIN, D, 1, 1, 0, 0, 0, 0, 0, 0, S, EPI_EMBED);

    for (int l = 0; l < L; ++l) {
        // pre-LN 1
        k_layernorm<<<nblk(BS, 8), 256, 0, stream>>>(h, ln1_g + (long)l * D, ln1_b + (long)l * D, y, (int)BS);
        // Q/K/V projections (f16 outputs)
        gemm(stream, y, wqT[l], bq + (long)l * D, nullptr, q, nullptr, nullptr, nullptr,
             (int)BS, D, D, D, D, D, 1, 1, 0, 0, 0, 0, 0, 0, S, EPI_BIAS_H);
        gemm(stream, y, wkT[l], bk + (long)l * D, nullptr, kk, nullptr, nullptr, nullptr,
             (int)BS, D, D, D, D, D, 1, 1, 0, 0, 0, 0, 0, 0, S, EPI_BIAS_H);
        gemm(stream, y, wvT[l], bv + (long)l * D, nullptr, v, nullptr, nullptr, nullptr,
             (int)BS, D, D, D, D, D, 1, 1, 0, 0, 0, 0, 0, 0, S, EPI_BIAS_H);
        // dd = (q*norm) @ proj^T  — batched over (b,h); norm folded into projh
        gemm(stream, q, projh[l], nullptr, dd, nullptr, nullptr, nullptr, nullptr,
             S, M, DH, D, DH, M, B * H, H,
             (long)S * D, DH, 0, 0, (long)H * S * M, (long)S * M, S, EPI_F32);
        k_qp<<<nblk(BHS, 256), 256, 0, stream>>>(dd, q, qp, BHS);
        // dd = (k*norm) @ proj^T
        gemm(stream, kk, projh[l], nullptr, dd, nullptr, nullptr, nullptr, nullptr,
             S, M, DH, D, DH, M, B * H, H,
             (long)S * D, DH, 0, 0, (long)H * S * M, (long)S * M, S, EPI_F32);
        k_rowmax<<<nblk(BHS, 256), 256, 0, stream>>>(dd, rmax, BHS);
        k_stab<<<dim3(B * H), 256, 0, stream>>>(rmax, stab);
        k_kp<<<nblk(BHS, 256), 256, 0, stream>>>(dd, kk, stab, kp, BHS);
        k_ksum<<<nblk(B * H * M, 256), 256, 0, stream>>>(kp, ksum, B * H * M);
        k_ctx<<<nblk((long)B * H * DH * M, 256), 256, 0, stream>>>(kp, v, ctxT, B * H * DH * M);
        k_dinv<<<nblk(BHS, 256), 256, 0, stream>>>(qp, ksum, dinv, BHS);
        // out = (qp @ ctx) * d_inv  -> att (f16, [BS x D] head-interleaved)
        gemm(stream, qp, ctxT, nullptr, nullptr, att, nullptr, nullptr, dinv,
             S, DH, M, M, M, D, B * H, H,
             (long)H * S * M, (long)S * M, (long)H * DH * M, (long)DH * M,
             (long)S * D, DH, S, EPI_DINV_H);
        // h += att @ wo + bo
        gemm(stream, att, woT[l], bo + (long)l * D, h, nullptr, h, nullptr, nullptr,
             (int)BS, D, D, D, D, D, 1, 1, 0, 0, 0, 0, 0, 0, S, EPI_RESID);
        // pre-LN 2 + FFN
        k_layernorm<<<nblk(BS, 8), 256, 0, stream>>>(h, ln2_g + (long)l * D, ln2_b + (long)l * D, y, (int)BS);
        gemm(stream, y, w1T[l], b1 + (long)l * FF, nullptr, ffn1, nullptr, nullptr, nullptr,
             (int)BS, FF, D, D, D, FF, 1, 1, 0, 0, 0, 0, 0, 0, S, EPI_GELU_H);
        gemm(stream, ffn1, w2T[l], b2 + (long)l * D, h, nullptr, h, nullptr, nullptr,
             (int)BS, D, FF, FF, FF, D, 1, 1, 0, 0, 0, 0, 0, 0, S, EPI_RESID);
    }

    // ---- mean pool over S, then final fc ----
    k_hbar<<<nblk(B * D, 256), 256, 0, stream>>>(h, hbar, B * D);
    k_fc<<<nblk(B * C, 128), 128, 0, stream>>>(hbar, fc_w, fc_b, out, B * C);
}
